// ObsEncoder_49529562858352
// MI455X (gfx1250) — matrix operand order repaired and checked by reading, not compiled, not run on hardware
//
#include <hip/hip_runtime.h>
#include <math.h>

typedef __attribute__((ext_vector_type(16))) __bf16 v16bf;
typedef __attribute__((ext_vector_type(8)))  __bf16 v8bf;
typedef __attribute__((ext_vector_type(8)))  float  v8f;
typedef __attribute__((ext_vector_type(4)))  float  v4f;

constexpr int HID  = 256;   // hidden width
constexpr int OBSD = 464;   // observation width
constexpr int BT   = 16;    // batch rows per workgroup (one WMMA M tile)

// ---------- fast device helpers ----------

// Hardware tanh (TRANS32 op on CDNA5) with branch-free fallback.
static __device__ __forceinline__ float fast_tanh(float x) {
#if __has_builtin(__builtin_amdgcn_tanhf)
  return __builtin_amdgcn_tanhf(x);
#elif __has_builtin(__builtin_amdgcn_tanh_f32)
  return __builtin_amdgcn_tanh_f32(x);
#else
  const float e = __expf(2.0f * x);                    // v_exp_f32 fast path
  return 1.0f - 2.0f * __builtin_amdgcn_rcpf(e + 1.0f);
#endif
}

static __device__ __forceinline__ void wait_async0() {
#if __has_builtin(__builtin_amdgcn_s_wait_asynccnt)
  __builtin_amdgcn_s_wait_asynccnt(0);
#else
  asm volatile("s_wait_asynccnt 0x0" ::: "memory");
#endif
}

// Async DMA: 16 bytes global -> LDS per active lane (ASYNCcnt tracked).
static __device__ __forceinline__ void async_ld16(unsigned lds_off, unsigned goff,
                                                  unsigned long long gbase) {
  asm volatile("global_load_async_to_lds_b128 %0, %1, %2"
               :: "v"(lds_off), "v"(goff), "s"(gbase) : "memory");
}

// ---------- WMMA helpers ----------

static __device__ __forceinline__ v8f wmma_bf16(v16bf a, v16bf b, v8f c) {
  // D = A(16x32 bf16) * B(32x16 bf16) + C(16x16 f32)
  return __builtin_amdgcn_wmma_f32_16x16x32_bf16(
      /*neg_a=*/false, a, /*neg_b=*/false, b,
      /*c_mod=*/(short)0, c, /*reuse_a=*/false, /*reuse_b=*/false);
}

// A fragment (16x32 bf16) gathered from a row-major LDS tile [16][ldk].
// lanes 0-15: M=lane, elems 0..15 -> K=kt*32+0..15;
// lanes 16-31: same +16 (matches prep_weight B layout).
static __device__ __forceinline__ v16bf lds_a_frag(const __bf16* sA, int ldk,
                                                   int kt, int lane) {
  const int m    = lane & 15;
  const int koff = kt * 32 + ((lane & 16) ? 16 : 0);
  const __bf16* r = sA + m * ldk + koff;
  v8bf lo = *(const v8bf*)(r);        // 16B ds_load
  v8bf hi = *(const v8bf*)(r + 8);    // 16B ds_load
  return __builtin_shufflevector(lo, hi, 0,1,2,3,4,5,6,7,8,9,10,11,12,13,14,15);
}

// B fragment: pre-swizzled in d_ws so each lane reads 32 contiguous bytes.
static __device__ __forceinline__ v16bf g_b_frag(const __bf16* blk, int lane) {
  return *(const v16bf*)(blk + lane * 16);
}

// ---------- weight pre-swizzle (fp32 [K][256] -> bf16 WMMA B fragments) ----------
// One wave per (kt,ct) block. Block = 32(K) x 16(N) bf16 = 512 elems.
// Lane l holds elems e=0..15 -> K = kt*32 + (l>=16 ? 16:0) + e, N = ct*16 + l%16.
__global__ void prep_weight(const float* __restrict__ src,
                            __bf16* __restrict__ dst, int K) {
  const int lane  = threadIdx.x;
  const int blk   = blockIdx.x;
  const int kt    = blk >> 4;
  const int ct    = blk & 15;
  const int n     = ct * 16 + (lane & 15);
  const int kbase = kt * 32 + ((lane & 16) ? 16 : 0);
  __bf16* d = dst + (size_t)blk * 512 + lane * 16;
#pragma unroll
  for (int e = 0; e < 16; ++e) {
    const int k = kbase + e;
    const float v = (k < K) ? src[(size_t)k * HID + n] : 0.0f;
    d[e] = (__bf16)v;
  }
}

// ---------- fused observation encoder ----------
// 1 workgroup = 16 batch rows, 4 wave32; wave w owns column tiles 4w..4w+3.
__global__ __launch_bounds__(128) void obs_encoder(
    const float* __restrict__ in,
    const float* __restrict__ b_self, const float* __restrict__ b_other,
    const float* __restrict__ b_box,  const float* __restrict__ b_ramp,
    const float* __restrict__ b_fc,   const float* __restrict__ b_e1,
    const float* __restrict__ b_e2,
    const __bf16* __restrict__ ws,
    float* __restrict__ out) {
  extern __shared__ char smem[];
  float*  sObs   = (float*)smem;                  // 16 x 464 f32      (29 KB)
  __bf16* sAself = (__bf16*)(sObs + 16 * OBSD);   // 16 x 32           (1 KB)
  __bf16* sEmbS  = sAself + 16 * 32;              // 16 x 256          (8 KB)
  __bf16* sCat   = sEmbS  + 16 * HID;             // 16 x 1024         (32 KB)
  __bf16* sH     = sCat   + 16 * 1024;            // 16 x 256          (8 KB)
  __bf16* sAent  = sH     + 16 * HID;             // 256 x 32          (16 KB)
  __bf16* sEmbE  = sAent  + 256 * 32;             // 256 x 256         (128 KB)
  float*  sQ     = (float*)(sEmbE + 256 * HID);   // 16 x 256 f32      (16 KB)
  float*  sBeta  = sQ + 16 * HID;                 // 16 x 16 f32       (1 KB)

  const int tid  = threadIdx.x;
  const int wave = tid >> 5;
  const int lane = tid & 31;
  const long rowbase = (long)blockIdx.x * BT;

  // weight regions (bf16 elements) inside d_ws
  const __bf16* Wself  = ws;                // 16 blocks
  const __bf16* Wother = ws + 8192;
  const __bf16* Wbox   = ws + 16384;
  const __bf16* Wramp  = ws + 24576;
  const __bf16* Cother = ws + 32768;        // 128 blocks each
  const __bf16* Cbox   = ws + 98304;
  const __bf16* Cramp  = ws + 163840;
  const __bf16* Wfc    = ws + 229376;
  const __bf16* We1    = ws + 294912;       // 512 blocks
  const __bf16* We2    = ws + 557056;

  // ---- Phase A0: async-DMA the whole obs tile (16 x 1856 B) into LDS ----
  {
    const unsigned long long gbase = (unsigned long long)(uintptr_t)in;
    const unsigned lbase = (unsigned)(uintptr_t)(void*)sObs;
    for (int t = tid; t < 16 * 116; t += 128) {   // 116 = 1856B/row / 16B
      const int r = t / 116, c = t - r * 116;
      const unsigned goff = (unsigned)((rowbase + r) * OBSD * 4 + c * 16);
      const unsigned loff = lbase + (unsigned)(r * OBSD * 4 + c * 16);
      async_ld16(loff, goff, gbase);
    }
    wait_async0();
  }
  __syncthreads();

  // ---- Phase A1: stage self features, zero-padded K 10->32 ----
  for (int idx = tid; idx < 16 * 32; idx += 128) {
    const int r = idx >> 5, k = idx & 31;
    const float v = (k < 10) ? sObs[r * OBSD + k] : 0.0f;
    sAself[idx] = (__bf16)v;
  }
  __syncthreads();

  // ---- Phase B: emb_self = tanh(x_self @ W_self + b) ----
  {
    const v16bf a = lds_a_frag(sAself, 32, 0, lane);
    const int mb = (lane & 16) ? 8 : 0;
#pragma unroll
    for (int i = 0; i < 4; ++i) {
      const int ct = wave * 4 + i;
      v8f acc = {};
      acc = wmma_bf16(a, g_b_frag(Wself + ct * 512, lane), acc);
      const int n = ct * 16 + (lane & 15);
      const float bv = b_self[n];
#pragma unroll
      for (int j = 0; j < 8; ++j)
        sEmbS[(mb + j) * HID + n] = (__bf16)fast_tanh(acc[j] + bv);
    }
  }
  __syncthreads();

  // ---- Phase C: gi = tanh(emb_self @ W_fc + b) -> sCat[:, 0:256] ----
  {
    v8f acc[4];
#pragma unroll
    for (int i = 0; i < 4; ++i) acc[i] = (v8f){};
    for (int kt = 0; kt < 8; ++kt) {
      const v16bf a = lds_a_frag(sEmbS, HID, kt, lane);
#pragma unroll
      for (int i = 0; i < 4; ++i) {
        const int ct = wave * 4 + i;
        acc[i] = wmma_bf16(a, g_b_frag(Wfc + (kt * 16 + ct) * 512, lane), acc[i]);
      }
    }
    const int mb = (lane & 16) ? 8 : 0;
#pragma unroll
    for (int i = 0; i < 4; ++i) {
      const int ct = wave * 4 + i;
      const int n = ct * 16 + (lane & 15);
      const float bv = b_fc[n];
#pragma unroll
      for (int j = 0; j < 8; ++j)
        sCat[(mb + j) * 1024 + n] = (__bf16)fast_tanh(acc[i][j] + bv);
    }
  }

  // ---- Phase D: three attention-pool groups ----
  const int   NeA[3]  = {15, 16, 8};
  const int   KgA[3]  = {10, 13, 12};
  const int   offA[3] = {10, 160, 368};
  const __bf16* WgA[3] = {Wother, Wbox, Wramp};
  const __bf16* CgA[3] = {Cother, Cbox, Cramp};
  const float*  bgA[3] = {b_other, b_box, b_ramp};

  for (int g = 0; g < 3; ++g) {
    const int Ne = NeA[g], Kg = KgA[g], base = offA[g];
    __syncthreads();  // protect reuse of sAent / sEmbE / sQ across groups

    // D1: stage entity features; LDS row = n*16 + r (tile rt == entity n)
    for (int idx = tid; idx < Ne * 16 * 32; idx += 128) {
      const int n = idx >> 9;
      const int r = (idx >> 5) & 15;
      const int k = idx & 31;
      const float v = (k < Kg) ? sObs[r * OBSD + base + n * Kg + k] : 0.0f;
      sAent[idx] = (__bf16)v;
    }
    __syncthreads();

    // D2: entity embeddings emb[n] = tanh(ents @ Wg + bg)
    for (int rt = 0; rt < Ne; ++rt) {
      const v16bf a = lds_a_frag(sAent + rt * 16 * 32, 32, 0, lane);
      const int mb = (lane & 16) ? 8 : 0;
#pragma unroll
      for (int i = 0; i < 4; ++i) {
        const int ct = wave * 4 + i;
        v8f acc = {};
        acc = wmma_bf16(a, g_b_frag(WgA[g] + ct * 512, lane), acc);
        const int n = ct * 16 + (lane & 15);
        const float bv = bgA[g][n];
#pragma unroll
        for (int j = 0; j < 8; ++j)
          sEmbE[(rt * 16 + mb + j) * HID + n] = (__bf16)fast_tanh(acc[j] + bv);
      }
    }

    // D3: q = emb_self @ corr_g  (kept f32)
    {
      v8f acc[4];
#pragma unroll
      for (int i = 0; i < 4; ++i) acc[i] = (v8f){};
      for (int kt = 0; kt < 8; ++kt) {
        const v16bf a = lds_a_frag(sEmbS, HID, kt, lane);
#pragma unroll
        for (int i = 0; i < 4; ++i) {
          const int ct = wave * 4 + i;
          acc[i] = wmma_bf16(a, g_b_frag(CgA[g] + (kt * 16 + ct) * 512, lane), acc[i]);
        }
      }
      const int mb = (lane & 16) ? 8 : 0;
#pragma unroll
      for (int i = 0; i < 4; ++i) {
        const int ct = wave * 4 + i;
        const int n = ct * 16 + (lane & 15);
#pragma unroll
        for (int j = 0; j < 8; ++j) sQ[(mb + j) * HID + n] = acc[i][j];
      }
    }
    __syncthreads();

    // D4: beta[r][n] = q[r] . emb[n][r]   (vectorized LDS dot)
    for (int p = tid; p < 16 * Ne; p += 128) {
      const int r = p & 15, n = p >> 4;
      const float* q = sQ + r * HID;
      const __bf16* e = sEmbE + (n * 16 + r) * HID;
      float s = 0.f;
      for (int h = 0; h < HID; h += 8) {
        const v4f  q0 = *(const v4f*)(q + h);
        const v4f  q1 = *(const v4f*)(q + h + 4);
        const v8bf ev = *(const v8bf*)(e + h);
#pragma unroll
        for (int u = 0; u < 4; ++u) s += q0[u] * (float)ev[u];
#pragma unroll
        for (int u = 0; u < 4; ++u) s += q1[u] * (float)ev[4 + u];
      }
      sBeta[r * 16 + n] = s;
    }
    __syncthreads();

    // D5a: softmax over entities (one thread per batch row)
    if (tid < 16) {
      const int r = tid;
      float mx = -3.0e38f;
      for (int n = 0; n < Ne; ++n) mx = fmaxf(mx, sBeta[r * 16 + n]);
      float ssum = 0.f;
      for (int n = 0; n < Ne; ++n) {
        const float e = __expf(sBeta[r * 16 + n] - mx);
        sBeta[r * 16 + n] = e;
        ssum += e;
      }
      const float inv = 1.0f / ssum;
      for (int n = 0; n < Ne; ++n) sBeta[r * 16 + n] *= inv;
    }
    __syncthreads();

    // D5b: vi[r] = sum_n alpha[r][n] * emb[n][r] -> sCat[:, 256*(g+1)..]
    const int catoff = 256 * (g + 1);
    for (int c = tid; c < 16 * (HID / 8); c += 128) {
      const int r  = c >> 5;
      const int h0 = (c & 31) * 8;
      float a8[8];
#pragma unroll
      for (int u = 0; u < 8; ++u) a8[u] = 0.f;
      for (int n = 0; n < Ne; ++n) {
        const float al = sBeta[r * 16 + n];
        const v8bf ev = *(const v8bf*)(sEmbE + (n * 16 + r) * HID + h0);
#pragma unroll
        for (int u = 0; u < 8; ++u) a8[u] += al * (float)ev[u];
      }
      v8bf ov;
#pragma unroll
      for (int u = 0; u < 8; ++u) ov[u] = (__bf16)a8[u];
      *(v8bf*)(sCat + r * 1024 + catoff + h0) = ov;
    }
  }
  __syncthreads();

  // ---- Phase E: h = tanh(cat(16x1024) @ W_e1 + b) ----
  {
    v8f acc[4];
#pragma unroll
    for (int i = 0; i < 4; ++i) acc[i] = (v8f){};
    for (int kt = 0; kt < 32; ++kt) {
      const v16bf a = lds_a_frag(sCat, 1024, kt, lane);
#pragma unroll
      for (int i = 0; i < 4; ++i) {
        const int ct = wave * 4 + i;
        acc[i] = wmma_bf16(a, g_b_frag(We1 + (kt * 16 + ct) * 512, lane), acc[i]);
      }
    }
    const int mb = (lane & 16) ? 8 : 0;
#pragma unroll
    for (int i = 0; i < 4; ++i) {
      const int ct = wave * 4 + i;
      const int n = ct * 16 + (lane & 15);
      const float bv = b_e1[n];
#pragma unroll
      for (int j = 0; j < 8; ++j)
        sH[(mb + j) * HID + n] = (__bf16)fast_tanh(acc[i][j] + bv);
    }
  }
  __syncthreads();

  // ---- Phase F: out = tanh(h @ W_e2 + b) -> global f32 ----
  {
    v8f acc[4];
#pragma unroll
    for (int i = 0; i < 4; ++i) acc[i] = (v8f){};
    for (int kt = 0; kt < 8; ++kt) {
      const v16bf a = lds_a_frag(sH, HID, kt, lane);
#pragma unroll
      for (int i = 0; i < 4; ++i) {
        const int ct = wave * 4 + i;
        acc[i] = wmma_bf16(a, g_b_frag(We2 + (kt * 16 + ct) * 512, lane), acc[i]);
      }
    }
    const int mb = (lane & 16) ? 8 : 0;
#pragma unroll
    for (int i = 0; i < 4; ++i) {
      const int ct = wave * 4 + i;
      const int n = ct * 16 + (lane & 15);
      const float bv = b_e2[n];
#pragma unroll
      for (int j = 0; j < 8; ++j)
        out[(rowbase + mb + j) * HID + n] = fast_tanh(acc[i][j] + bv);
    }
  }
}

extern "C" void kernel_launch(void* const* d_in, const int* in_sizes, int n_in,
                              void* d_out, int out_size, void* d_ws, size_t ws_size,
                              hipStream_t stream) {
  (void)n_in; (void)out_size; (void)ws_size;
  const float* in = (const float*)d_in[0];
  __bf16* ws = (__bf16*)d_ws;

  // Pre-swizzle all weights to bf16 WMMA B-fragment layout in d_ws (~1.2 MB).
  prep_weight<<<16,  32, 0, stream>>>((const float*)d_in[1],  ws,          10);   // W_self
  prep_weight<<<16,  32, 0, stream>>>((const float*)d_in[3],  ws + 8192,   10);   // W_other
  prep_weight<<<16,  32, 0, stream>>>((const float*)d_in[5],  ws + 16384,  13);   // W_box
  prep_weight<<<16,  32, 0, stream>>>((const float*)d_in[7],  ws + 24576,  12);   // W_ramp
  prep_weight<<<128, 32, 0, stream>>>((const float*)d_in[9],  ws + 32768,  256);  // corr_other
  prep_weight<<<128, 32, 0, stream>>>((const float*)d_in[10], ws + 98304,  256);  // corr_box
  prep_weight<<<128, 32, 0, stream>>>((const float*)d_in[11], ws + 163840, 256);  // corr_ramp
  prep_weight<<<128, 32, 0, stream>>>((const float*)d_in[12], ws + 229376, 256);  // W_fc
  prep_weight<<<512, 32, 0, stream>>>((const float*)d_in[14], ws + 294912, 1024); // W_e1
  prep_weight<<<128, 32, 0, stream>>>((const float*)d_in[16], ws + 557056, 256);  // W_e2

  const int rows = in_sizes[0] / OBSD;         // 32768
  const int grid = rows / BT;                  // 2048 workgroups
  const int smem = 244736;                     // 239 KB dynamic LDS (WGP has 320 KB)
  hipFuncSetAttribute((const void*)obs_encoder,
                      hipFuncAttributeMaxDynamicSharedMemorySize, smem);
  obs_encoder<<<grid, 128, smem, stream>>>(
      in,
      (const float*)d_in[2],  (const float*)d_in[4],
      (const float*)d_in[6],  (const float*)d_in[8],
      (const float*)d_in[13], (const float*)d_in[15],
      (const float*)d_in[17],
      ws, (float*)d_out);
}
